// Model_2705829396665
// MI455X (gfx1250) — matrix operand order repaired and checked by reading, not compiled, not run on hardware
//
#include <hip/hip_runtime.h>
#include <hip/hip_bf16.h>

typedef __attribute__((ext_vector_type(16))) __bf16 v16bf;
typedef __attribute__((ext_vector_type(8)))  __bf16 v8bf;
typedef __attribute__((ext_vector_type(8)))  float  v8f;

#define ACT_NONE     0
#define ACT_SILU     1
#define ACT_GELU     2
#define ACT_SOFTPLUS 3

#define LDK 40   // LDS K-stride (bf16): 80B rows keep every 16B quarter aligned

__device__ __forceinline__ float siluf(float x) { return x / (1.f + __expf(-x)); }

// async 16B global->LDS copy (CDNA5 GLOBAL_LOAD_ASYNC_TO_LDS_B128, ASYNCcnt)
__device__ __forceinline__ void async_b128(unsigned lds_off, const __bf16* gp) {
  asm volatile("global_load_async_to_lds_b128 %0, %1, off"
               :: "v"(lds_off), "v"((unsigned long long)(size_t)gp) : "memory");
}
// wait until at most `n` async ops outstanding (loads complete in order, so
// waiting <=2 with 2 just-issued copies guarantees the previous tile is done)
__device__ __forceinline__ void wait_async0() {
  asm volatile("s_wait_asynccnt 0x0" ::: "memory");
}
__device__ __forceinline__ void wait_async2() {
  asm volatile("s_wait_asynccnt 0x2" ::: "memory");
}

// ---------------------------------------------------------------------------
// WMMA GEMM: C[m,n] = epi( sum_k A[m,k] * W[n,k] )
// A: MxK f32 (row stride lda, even).  W: bf16 arena, rows padded to the launch
// grid (gridDim.x*128) and cols zero-padded to ldw = roundup(K,32).
// Block tile 64x128, 8 waves (2x4), wave tile 32x32 = 2x2 wmma frags.
// Double-buffered LDS: tile i+1's async-to-LDS copies overlap tile i's WMMAs,
// synchronized with partial ASYNCcnt waits.
// ---------------------------------------------------------------------------
__global__ __launch_bounds__(256) void k_gemm_bf16(
    const float* __restrict__ A, int lda,
    const __bf16* __restrict__ W, int ldw,
    const float* __restrict__ bias,
    const float* __restrict__ res,
    float* __restrict__ C, int ldc,
    int M, int N, int K, int act)
{
  __shared__ __align__(16) __bf16 As[2][64 * LDK];
  __shared__ __align__(16) __bf16 Bs[2][128 * LDK];

  const int tid  = threadIdx.x;
  const int mb   = blockIdx.y * 64;
  const int nb   = blockIdx.x * 128;
  const int wave = tid >> 5;
  const int lane = tid & 31;
  const int wm   = (wave >> 2) * 32;
  const int wn   = (wave & 3) * 32;
  const int l15  = lane & 15;
  const int h    = lane >> 4;

  // staging geometry: one 8-elem (16B) unit = (row, quarter)
  const int ar = tid >> 2;            // row 0..63
  const int aq = tid & 3;             // quarter 0..3
  const bool mfull = (mb + 64 <= M);

  auto stageB = [&](int buf, int k0) {
    const __bf16* g0 = W + (long)(nb + ar)      * ldw + k0 + aq * 8;
    const __bf16* g1 = W + (long)(nb + 64 + ar) * ldw + k0 + aq * 8;
    async_b128((unsigned)(size_t)&Bs[buf][(ar)      * LDK + aq * 8], g0);
    async_b128((unsigned)(size_t)&Bs[buf][(64 + ar) * LDK + aq * 8], g1);
  };
  auto stageA = [&](int buf, int k0) {
    if (mfull && (k0 + 32 <= K)) {
      const float* g = A + (long)(mb + ar) * lda + k0 + aq * 8;
      float f[8];
      const float2 p0 = *(const float2*)(g + 0);
      const float2 p1 = *(const float2*)(g + 2);
      const float2 p2 = *(const float2*)(g + 4);
      const float2 p3 = *(const float2*)(g + 6);
      f[0] = p0.x; f[1] = p0.y; f[2] = p1.x; f[3] = p1.y;
      f[4] = p2.x; f[5] = p2.y; f[6] = p3.x; f[7] = p3.y;
      union { v8bf v; __bf16 e[8]; } u;
#pragma unroll
      for (int i = 0; i < 8; ++i) u.e[i] = (__bf16)f[i];
      *(v8bf*)&As[buf][ar * LDK + aq * 8] = u.v;
      if (k0 + 32 < K) __builtin_prefetch(g + 32, 0, 3);
    } else {
      // edge tile: guarded scalar staging
#pragma unroll
      for (int i = 0; i < 8; ++i) {
        int e = i * 256 + tid;
        int r = e >> 5, c = e & 31;
        int m = mb + r, k = k0 + c;
        float v = (m < M && k < K) ? A[(long)m * lda + k] : 0.f;
        As[buf][r * LDK + c] = (__bf16)v;
      }
    }
  };

  v8f acc[2][2] = {};
  const int niter = (K + 31) / 32;

  stageB(0, 0);
  stageA(0, 0);
  for (int i = 0; i < niter; ++i) {
    const int cur = i & 1, nxt = cur ^ 1;
    if (i + 1 < niter) {               // prefetch next tile into other buffer
      stageB(nxt, (i + 1) * 32);
      stageA(nxt, (i + 1) * 32);
      wait_async2();                   // tile i complete; tile i+1 in flight
    } else {
      wait_async0();
    }
    __syncthreads();                   // tile i visible to all waves

    union Frag { v16bf v; v8bf q[2]; } afr[2], bfr[2];
    // A 16x32 bf16 frag: half h -> K chunks [8h, 8h+7], [16+8h, 16+8h+7]
#pragma unroll
    for (int ii = 0; ii < 2; ++ii) {
      int row = wm + ii * 16 + l15;
      afr[ii].q[0] = *(const v8bf*)&As[cur][row * LDK + h * 8];
      afr[ii].q[1] = *(const v8bf*)&As[cur][row * LDK + 16 + h * 8];
    }
    // B 32x16 frag (column-major): lane = N column, half h -> K chunks [8h, 8h+7], [16+8h, 16+8h+7]
#pragma unroll
    for (int j = 0; j < 2; ++j) {
      int row = wn + j * 16 + l15;
      bfr[j].q[0] = *(const v8bf*)&Bs[cur][row * LDK + h * 8];
      bfr[j].q[1] = *(const v8bf*)&Bs[cur][row * LDK + 16 + h * 8];
    }
#pragma unroll
    for (int ii = 0; ii < 2; ++ii)
#pragma unroll
      for (int j = 0; j < 2; ++j)
        acc[ii][j] = __builtin_amdgcn_wmma_f32_16x16x32_bf16(
            false, afr[ii].v, false, bfr[j].v, (short)0, acc[ii][j], false, false);
    __syncthreads();                   // all reads of buf[cur] done before restage
  }

  // epilogue: C/D layout -> lane n = l15, VGPR r -> m = r + 8h
#pragma unroll
  for (int i = 0; i < 2; ++i)
#pragma unroll
    for (int j = 0; j < 2; ++j) {
      int n = nb + wn + j * 16 + l15;
      float bv = (bias != nullptr && n < N) ? bias[n] : 0.f;
#pragma unroll
      for (int r = 0; r < 8; ++r) {
        int m = mb + wm + i * 16 + r + 8 * h;
        if (m < M && n < N) {
          float t = acc[i][j][r] + bv;
          if (act == ACT_SILU)          t = siluf(t);
          else if (act == ACT_GELU)     t = 0.5f * t * (1.f + erff(t * 0.70710678f));
          else if (act == ACT_SOFTPLUS) t = (t > 20.f) ? t : log1pf(__expf(t));
          if (res != nullptr) t = res[(long)m * ldc + n] - t;
          C[(long)m * ldc + n] = t;
        }
      }
    }
}

// f32 weight (N x K) -> padded bf16 arena (Np x Kp), zero fill OOB
__global__ void k_cvt_w(const float* __restrict__ w, __bf16* __restrict__ out,
                        int N, int K, int Kp, long total)
{
  long idx = (long)blockIdx.x * blockDim.x + threadIdx.x;
  if (idx >= total) return;
  int n = (int)(idx / Kp), k = (int)(idx % Kp);
  float v = (n < N && k < K) ? w[(long)n * K + k] : 0.f;
  out[idx] = (__bf16)v;
}

// rfft basis -> padded bf16 arena: rows j<df: cos / -sin (ortho), else 0
__global__ void k_gen_rfft_bf(__bf16* __restrict__ Wr, int d, int Kp, long total)
{
  int nf = d / 2 + 1, df = 2 * nf;
  long idx = (long)blockIdx.x * blockDim.x + threadIdx.x;
  if (idx >= total) return;
  int j = (int)(idx / Kp), n = (int)(idx % Kp);
  float val = 0.f;
  if (j < df && n < d) {
    int k = (j < nf) ? j : j - nf;
    float scale = rsqrtf((float)d);
    float ang = 6.28318530718f * (float)((k * n) % d) / (float)d;
    val = (j < nf) ? cosf(ang) * scale : -sinf(ang) * scale;
  }
  Wr[idx] = (__bf16)val;
}

// irfft basis -> padded bf16 arena (rows n < d, cols j < df)
__global__ void k_gen_irfft_bf(__bf16* __restrict__ Wi, int d, int Kp, long total)
{
  int nf = d / 2 + 1, df = 2 * nf;
  long idx = (long)blockIdx.x * blockDim.x + threadIdx.x;
  if (idx >= total) return;
  int n = (int)(idx / Kp), j = (int)(idx % Kp);
  float val = 0.f;
  if (n < d && j < df) {
    float scale = rsqrtf((float)d);
    if (j < nf) {
      int k = j;
      float c = (k == 0 || k == d / 2) ? 1.f : 2.f;
      float ang = 6.28318530718f * (float)((k * n) % d) / (float)d;
      val = c * cosf(ang) * scale;
    } else {
      int k = j - nf;
      if (k != 0 && k != d / 2) {
        float ang = 6.28318530718f * (float)((k * n) % d) / (float)d;
        val = -2.f * sinf(ang) * scale;
      }
    }
  }
  Wi[idx] = (__bf16)val;
}

// instance norm over seq + transposed write xn_t[(b*V+v), s]
__global__ void k_norm_stats(const float* __restrict__ xe, float* __restrict__ xn_t,
                             float* __restrict__ mean, float* __restrict__ stdv,
                             int B, int S, int V)
{
  int row = blockIdx.x;
  int b = row / V, v = row - b * V;
  const float* base = xe + (long)b * S * V + v;
  float s1 = 0.f, s2 = 0.f;
  for (int s = threadIdx.x; s < S; s += 256) {
    float x = base[(long)s * V];
    s1 += x; s2 += x * x;
  }
  __shared__ float ra[256], rb[256];
  ra[threadIdx.x] = s1; rb[threadIdx.x] = s2;
  __syncthreads();
  for (int st = 128; st > 0; st >>= 1) {
    if (threadIdx.x < st) { ra[threadIdx.x] += ra[threadIdx.x + st]; rb[threadIdx.x] += rb[threadIdx.x + st]; }
    __syncthreads();
  }
  float mu  = ra[0] / S;
  float var = rb[0] / S - mu * mu;
  float sd  = sqrtf(var + 1e-5f);
  if (threadIdx.x == 0) { mean[row] = mu; stdv[row] = sd; }
  float inv = 1.f / sd;
  for (int s = threadIdx.x; s < S; s += 256)
    xn_t[(long)row * S + s] = (base[(long)s * V] - mu) * inv;
}

// LayerNorm: t = in (+in2); if(pre_silu) silu; out = LN(t)*g+b (+res)
__global__ void k_layernorm(const float* __restrict__ in, const float* __restrict__ in2,
                            const float* __restrict__ g, const float* __restrict__ bb,
                            const float* __restrict__ res, float* __restrict__ out,
                            int d, int pre_silu)
{
  long row = blockIdx.x;
  __shared__ float sbuf[544];
  __shared__ float red[256];
  const float* pi = in + row * d;
  const float* p2 = (in2 != nullptr) ? in2 + row * d : nullptr;
  float s1 = 0.f;
  for (int s = threadIdx.x; s < d; s += 256) {
    float t = pi[s] + (p2 ? p2[s] : 0.f);
    if (pre_silu) t = siluf(t);
    sbuf[s] = t; s1 += t;
  }
  red[threadIdx.x] = s1;
  __syncthreads();
  for (int st = 128; st > 0; st >>= 1) {
    if (threadIdx.x < st) red[threadIdx.x] += red[threadIdx.x + st];
    __syncthreads();
  }
  float mu = red[0] / d;
  __syncthreads();
  float s2 = 0.f;
  for (int s = threadIdx.x; s < d; s += 256) { float c = sbuf[s] - mu; s2 += c * c; }
  red[threadIdx.x] = s2;
  __syncthreads();
  for (int st = 128; st > 0; st >>= 1) {
    if (threadIdx.x < st) red[threadIdx.x] += red[threadIdx.x + st];
    __syncthreads();
  }
  float rstd = rsqrtf(red[0] / d + 1e-5f);
  for (int s = threadIdx.x; s < d; s += 256) {
    float o = (sbuf[s] - mu) * rstd * g[s] + bb[s];
    if (res != nullptr) o += res[row * d + s];
    out[row * d + s] = o;
  }
}

// depthwise causal conv (width 4) over token dim + SiLU
__global__ void k_conv_silu(const float* __restrict__ xz, int ldxz,
                            const float* __restrict__ cw, const float* __restrict__ cb,
                            float* __restrict__ xc, int di, int V, long total)
{
  long idx = (long)blockIdx.x * blockDim.x + threadIdx.x;
  if (idx >= total) return;
  int  c = (int)(idx % di);
  long m = idx / di;
  int  v = (int)(m % V);
  float acc = cb[c];
#pragma unroll
  for (int t = 0; t < 4; ++t) {
    int vv = v - 3 + t;
    if (vv >= 0) acc += xz[(m + (long)(vv - v)) * ldxz + c] * cw[c * 4 + t];
  }
  xc[m * di + c] = siluf(acc);
}

// selective scan: one thread per (batch, channel), 16 states in registers
__global__ void k_scan(const float* __restrict__ dt, const float* __restrict__ xc,
                       const float* __restrict__ dblb, int lddbl, int dtr,
                       const float* __restrict__ A_log, const float* __restrict__ Dp,
                       float* __restrict__ y, int di, int V, int B)
{
  int idx = blockIdx.x * blockDim.x + threadIdx.x;
  if (idx >= B * di) return;
  int ch = idx % di, b = idx / di;
  float a[16], h[16];
#pragma unroll
  for (int s = 0; s < 16; ++s) { a[s] = -__expf(A_log[ch * 16 + s]); h[s] = 0.f; }
  float Dv = Dp[ch];
  for (int v = 0; v < V; ++v) {
    long m = (long)b * V + v;
    float dtv = dt[m * di + ch];
    float xv  = xc[m * di + ch];
    const float* bc = dblb + m * lddbl + dtr;  // B[0..15], C[16..31]
    float dx = dtv * xv;
    float acc = 0.f;
#pragma unroll
    for (int s = 0; s < 16; ++s) {
      h[s] = __expf(dtv * a[s]) * h[s] + dx * bc[s];
      acc += h[s] * bc[16 + s];
    }
    y[m * di + ch] = acc + Dv * xv;
  }
}

// y *= silu(z), z = xz[:, di:]
__global__ void k_gate(float* __restrict__ y, const float* __restrict__ xz,
                       int ldxz, int di, long total)
{
  long idx = (long)blockIdx.x * blockDim.x + threadIdx.x;
  if (idx >= total) return;
  int  c = (int)(idx % di);
  long m = idx / di;
  y[idx] *= siluf(xz[m * ldxz + di + c]);
}

// de-normalize + transpose (b,v,t)->(b,t,v)
__global__ void k_final(const float* __restrict__ dec, const float* __restrict__ mean,
                        const float* __restrict__ stdv, float* __restrict__ out,
                        int B, int P, int V)
{
  long idx = (long)blockIdx.x * blockDim.x + threadIdx.x;
  if (idx >= (long)B * P * V) return;
  int v = (int)(idx % V);
  int t = (int)((idx / V) % P);
  int b = (int)(idx / ((long)V * P));
  long row = (long)b * V + v;
  out[idx] = dec[row * P + t] * stdv[row] + mean[row];
}

// ---------------------------------------------------------------------------
extern "C" void kernel_launch(void* const* d_in, const int* in_sizes, int n_in,
                              void* d_out, int out_size, void* d_ws, size_t ws_size,
                              hipStream_t stream)
{
  (void)in_sizes; (void)n_in; (void)out_size; (void)ws_size;
  const int B = 32, S = 512, V = 321, D = 512, P = 96;
  const int M = B * V;                 // 10272
  const int DF = 2 * (D / 2 + 1);      // 514
  const int LDB = 72;                  // padded dbl row stride (even, 16B rows)

  auto F = [&](int i) { return (const float*)d_in[i]; };
  // pytree leaf order: x_enc,x_mark_enc,x_dec,x_mark_dec, then params key-sorted:
  // emb_b(4), emb_w(5), layer l at 6+36*l: freq rl(17), ln_b(+17), ln_g(+18),
  // time rl(+19..); proj_b(78), proj_w(79).
  // rl: ff_b1,ff_b2,ff_w1,ff_w2,ln1_b,ln1_g,ln2_b,ln2_g,
  //     A_log,D,W_dt,W_in,W_out,W_x,b_dt,conv_b,conv_w

  char* wp = (char*)d_ws;
  auto allocF = [&](size_t n) {
    float* p = (float*)wp; wp += ((n * 4 + 255) / 256) * 256; return p;
  };
  auto allocB = [&](size_t n) {
    __bf16* p = (__bf16*)wp; wp += ((n * 2 + 255) / 256) * 256; return p;
  };

  float* mean  = allocF(M);
  float* stdv  = allocF(M);
  float* buf_a = allocF((size_t)M * 512);   // xn_t, later time-branch output
  float* x     = allocF((size_t)M * 512);
  float* xz    = allocF((size_t)M * 2056);
  float* xc    = allocF((size_t)M * 1028);
  float* dblb  = allocF((size_t)M * LDB);
  float* buf_b = allocF((size_t)M * 1028);  // dt, later FF hidden
  float* ybuf  = allocF((size_t)M * 1028);
  float* r1    = allocF((size_t)M * 514);
  float* r1n   = allocF((size_t)M * 514);
  float* r2    = allocF((size_t)M * 514);
  float* rlout = allocF((size_t)M * 514);
  float* spin  = allocF((size_t)M * 514);
  float* spec  = allocF((size_t)M * 512);
  float* dec   = allocF((size_t)M * P);

  // ---- bf16 weight arena (zero-padded: rows->mult of 128, cols->mult of 32)
  auto cvtW = [&](const float* w, int N, int K, int* ldw) {
    int Np = (N + 127) & ~127, Kp = (K + 31) & ~31;
    __bf16* p = allocB((size_t)Np * Kp);
    long tot = (long)Np * Kp;
    k_cvt_w<<<(int)((tot + 255) / 256), 256, 0, stream>>>(w, p, N, K, Kp, tot);
    *ldw = Kp;
    return p;
  };

  __bf16 *wWin[4], *wWx[4], *wWdt[4], *wWout[4], *wF1[4], *wF2[4];
  int     lWin[4],  lWx[4],  lWdt[4],  lWout[4],  lF1[4],  lF2[4];
  auto setupW = [&](int pb, int d, int wi) {
    int di = 2 * d, dtr = (d + 15) / 16;
    wF1[wi]   = cvtW(F(pb + 2),  2 * d,    d,   &lF1[wi]);
    wF2[wi]   = cvtW(F(pb + 3),  d,        2*d, &lF2[wi]);
    wWdt[wi]  = cvtW(F(pb + 10), di,       dtr, &lWdt[wi]);
    wWin[wi]  = cvtW(F(pb + 11), 2 * di,   d,   &lWin[wi]);
    wWout[wi] = cvtW(F(pb + 12), d,        di,  &lWout[wi]);
    wWx[wi]   = cvtW(F(pb + 13), dtr + 32, di,  &lWx[wi]);
  };
  for (int l = 0; l < 2; ++l) {
    int base = 6 + 36 * l;
    setupW(base,      DF, l * 2 + 1);   // freq
    setupW(base + 19, D,  l * 2 + 0);   // time
  }
  int lEmb, lProj;
  __bf16* wEmb  = cvtW(F(5),  D, S, &lEmb);
  __bf16* wProj = cvtW(F(79), P, D, &lProj);
  // DFT bases directly into padded bf16 arena
  int lWr = 512, lWi = 544;
  __bf16* wWr = allocB((size_t)640 * 512);  // Np(514)=640, Kp(512)=512
  __bf16* wWi = allocB((size_t)512 * 544);  // Np(512)=512, Kp(514)=544
  {
    long t1 = 640L * 512, t2 = 512L * 544;
    k_gen_rfft_bf<<<(int)((t1 + 255) / 256), 256, 0, stream>>>(wWr, D, lWr, t1);
    k_gen_irfft_bf<<<(int)((t2 + 255) / 256), 256, 0, stream>>>(wWi, D, lWi, t2);
  }

  auto gemm = [&](const float* Ap, int lda, const __bf16* Wp, int ldw,
                  int N, int K, const float* bias, const float* resp,
                  float* Cp, int ldc, int act) {
    dim3 g((N + 127) / 128, (M + 63) / 64);
    k_gemm_bf16<<<g, 256, 0, stream>>>(Ap, lda, Wp, ldw, bias, resp, Cp, ldc, M, N, K, act);
  };
  auto ln = [&](const float* in, const float* in2, const float* g, const float* b,
                const float* resp, float* out, int d, int pre_silu) {
    k_layernorm<<<M, 256, 0, stream>>>(in, in2, g, b, resp, out, d, pre_silu);
  };

  auto rl_block = [&](const float* xin, int d, int pb, int wi, float* out) {
    int di = 2 * d, dtr = (d + 15) / 16;
    long tMdi = (long)M * di;
    gemm(xin, d, wWin[wi], lWin[wi], 2 * di, d, nullptr, nullptr, xz, 2 * di, ACT_NONE);
    k_conv_silu<<<(int)((tMdi + 255) / 256), 256, 0, stream>>>(
        xz, 2 * di, F(pb + 16), F(pb + 15), xc, di, V, tMdi);
    gemm(xc, di, wWx[wi], lWx[wi], dtr + 32, di, nullptr, nullptr, dblb, LDB, ACT_NONE);
    gemm(dblb, LDB, wWdt[wi], lWdt[wi], di, dtr, F(pb + 14), nullptr, buf_b, di, ACT_SOFTPLUS);
    k_scan<<<(B * di + 255) / 256, 256, 0, stream>>>(
        buf_b, xc, dblb, LDB, dtr, F(pb + 8), F(pb + 9), ybuf, di, V, B);
    k_gate<<<(int)((tMdi + 255) / 256), 256, 0, stream>>>(ybuf, xz, 2 * di, di, tMdi);
    gemm(ybuf, di, wWout[wi], lWout[wi], d, di, nullptr, xin, r1, d, ACT_NONE);
    ln(r1, nullptr, F(pb + 5), F(pb + 4), nullptr, r1n, d, 0);
    gemm(r1n, d, wF1[wi], lF1[wi], 2 * d, d, F(pb + 0), nullptr, buf_b, 2 * d, ACT_GELU);
    gemm(buf_b, 2 * d, wF2[wi], lF2[wi], d, 2 * d, F(pb + 1), r1n, r2, d, ACT_NONE);
    ln(r2, nullptr, F(pb + 7), F(pb + 6), xin, out, d, 1);
  };

  // normalization + transposed embedding input
  k_norm_stats<<<M, 256, 0, stream>>>(F(0), buf_a, mean, stdv, B, S, V);
  // x = xn_t @ emb_w^T + emb_b
  gemm(buf_a, S, wEmb, lEmb, D, S, F(4), nullptr, x, D, ACT_NONE);

  for (int l = 0; l < 2; ++l) {
    int base = 6 + 36 * l;
    rl_block(x, D, base + 19, l * 2 + 0, buf_a);                    // time
    gemm(x, D, wWr, lWr, DF, D, nullptr, nullptr, spin, DF, ACT_NONE);   // rfft
    rl_block(spin, DF, base + 0, l * 2 + 1, rlout);                 // freq
    gemm(rlout, DF, wWi, lWi, D, DF, nullptr, nullptr, spec, D, ACT_NONE); // irfft
    ln(buf_a, spec, F(base + 18), F(base + 17), nullptr, x, D, 0);  // combine
  }

  gemm(x, D, wProj, lProj, P, D, F(78), nullptr, dec, P, ACT_NONE);
  k_final<<<(B * P * V + 255) / 256, 256, 0, stream>>>(
      dec, mean, stdv, (float*)d_out, B, P, V);
}
